// MAB_44057774522768
// MI455X (gfx1250) — matrix operand order repaired and checked by reading, not compiled, not run on hardware
//
#include <hip/hip_runtime.h>
#include <hip/hip_bf16.h>

// ---------------------------------------------------------------------------
// MAB (multi-head attention block) for MI455X / gfx1250.
// All matmuls on v_wmma_f32_16x16x32_bf16 (fp32 accumulate).
// B=8, SQ=SK=1024, D(Q/K/V)=512, H=8, dh=64 (compile-time, per reference).
// ---------------------------------------------------------------------------

typedef __bf16 bf16_t;
typedef __attribute__((ext_vector_type(16))) __bf16 v16bf;
typedef __attribute__((ext_vector_type(8)))  __bf16 v8bf;
typedef __attribute__((ext_vector_type(4)))  __bf16 v4bf;
typedef __attribute__((ext_vector_type(8)))  float  v8f;

#define WMMA_BF16(a, b, c)                                                  \
  __builtin_amdgcn_wmma_f32_16x16x32_bf16(false, (a), false, (b),           \
                                          (short)0, (c), false, false)

static __device__ __forceinline__ v16bf join16(v8bf lo, v8bf hi) {
  return __builtin_shufflevector(lo, hi, 0, 1, 2, 3, 4, 5, 6, 7,
                                 8, 9, 10, 11, 12, 13, 14, 15);
}

// ds_swizzle xor within 16-lane halves (SWAPX patterns: and=0x1f, or=0).
template <int XORM>
static __device__ __forceinline__ float swz_xor(float v) {
  return __builtin_bit_cast(
      float, __builtin_amdgcn_ds_swizzle(__builtin_bit_cast(int, v),
                                         (XORM << 10) | 0x1f));
}
static __device__ __forceinline__ float xmax16(float v) {
  v = fmaxf(v, swz_xor<1>(v));
  v = fmaxf(v, swz_xor<2>(v));
  v = fmaxf(v, swz_xor<4>(v));
  v = fmaxf(v, swz_xor<8>(v));
  return v;
}
static __device__ __forceinline__ float xsum16(float v) {
  v += swz_xor<1>(v);
  v += swz_xor<2>(v);
  v += swz_xor<4>(v);
  v += swz_xor<8>(v);
  return v;
}

// ---------------------------------------------------------------------------
// fp32 -> bf16 elementwise convert, 4-wide
// ---------------------------------------------------------------------------
__global__ __launch_bounds__(256) void cvt_f32_to_bf16(
    const float* __restrict__ x, bf16_t* __restrict__ y, int n4) {
  int i = blockIdx.x * blockDim.x + threadIdx.x;
  if (i < n4) {
    float4 f = ((const float4*)x)[i];
    v4bf o;
    o[0] = (bf16_t)f.x; o[1] = (bf16_t)f.y;
    o[2] = (bf16_t)f.z; o[3] = (bf16_t)f.w;
    ((v4bf*)y)[i] = o;
  }
}

// ---------------------------------------------------------------------------
// GEMM: Y = X @ W^T + bias, X [M,K] bf16 row-major, W [N,K] bf16 row-major.
// One wave computes a 16x64 tile (A-frag reuse across 4 B-frags).
// MODE 0: Yf = y, Ybf = bf16(y)                 (Q projection)
// MODE 1: Yf = resid + relu(y)                  (FFN epilogue, -> d_out)
// MODE 2: Ybf[(b,h,d,sk)] = bf16(y) transposed  (V projection)
// MODE 3: Ybf = bf16(y)                         (K projection)
// Grid: (M/16)*(N/64) waves, 4 waves/block.
// ---------------------------------------------------------------------------
template <int MODE>
__global__ __launch_bounds__(128) void gemm_bf16_wmma(
    const bf16_t* __restrict__ X, const bf16_t* __restrict__ W,
    const float* __restrict__ bias, const float* __restrict__ resid,
    float* __restrict__ Yf, bf16_t* __restrict__ Ybf,
    int M, int N, int K, int spanRows) {
  const int lane = threadIdx.x & 31;
  const int wave = threadIdx.x >> 5;
  const int nTiles = N >> 6;
  const int task = blockIdx.x * 4 + wave;
  const int m0 = (task / nTiles) << 4;
  const int n0 = (task % nTiles) << 6;
  const int row = lane & 15;
  const int hi = lane >> 4;  // half-wave select

  v8f acc[4] = {v8f{}, v8f{}, v8f{}, v8f{}};

  const bf16_t* aPtr = X + (size_t)(m0 + row) * K + hi * 16;
  const bf16_t* bPtr = W + (size_t)(n0 + row) * K + hi * 16;
  const size_t bStep = (size_t)16 * K;

  for (int k0 = 0; k0 < K; k0 += 32) {
    __builtin_prefetch(aPtr + 128, 0, 0);   // next k-tiles of A
    __builtin_prefetch(bPtr + 128, 0, 0);   // next k-tiles of B
    v16bf a = join16(*(const v8bf*)aPtr, *(const v8bf*)(aPtr + 8));
#pragma unroll
    for (int j = 0; j < 4; ++j) {
      v16bf b = *(const v16bf*)(bPtr + (size_t)j * bStep);
      acc[j] = WMMA_BF16(a, b, acc[j]);
    }
    aPtr += 32;
    bPtr += 32;
  }

  const int rbase = hi * 8;
#pragma unroll
  for (int j = 0; j < 4; ++j) {
    const int col = n0 + j * 16 + row;
    const float bz = bias[col];
#pragma unroll
    for (int r = 0; r < 8; ++r) {
      const int gr = m0 + rbase + r;
      const size_t idx = (size_t)gr * N + col;
      const float v = acc[j][r] + bz;
      if constexpr (MODE == 0) {
        Yf[idx] = v;
        Ybf[idx] = (bf16_t)v;
      } else if constexpr (MODE == 1) {
        Yf[idx] = resid[idx] + fmaxf(v, 0.0f);
      } else if constexpr (MODE == 2) {
        // transposed: [b][h][d][sk], h = col/64, d = col%64, b = gr/span
        const int bIdx = gr / spanRows;
        const int skIdx = gr - bIdx * spanRows;
        const size_t idxT =
            ((size_t)(bIdx * nTiles * 4 /* == N/16? no: H */) * 0 +
             (size_t)((bIdx * (N >> 6) + (col >> 6)) * 64 + (col & 63))) *
                spanRows +
            skIdx;
        Ybf[idxT] = (bf16_t)v;
      } else {  // MODE 3
        Ybf[idx] = (bf16_t)v;
      }
    }
  }
}

// ---------------------------------------------------------------------------
// Flash attention over head slices. One wave owns a 16-query tile of one
// (batch, head). Streams 32-key tiles: scores via 4 WMMA, online softmax
// (half-wave row reductions via ds_swizzle), P restaged via LDS into
// A-fragment layout, P@V via 4 WMMA on transposed-V (contiguous B-frags).
// Grid: B*H*(SQ/16) waves, 4 waves/block.
// ---------------------------------------------------------------------------
__global__ __launch_bounds__(128) void attention_flash(
    const bf16_t* __restrict__ Qbf, const float* __restrict__ Qf,
    const bf16_t* __restrict__ Kbf, const bf16_t* __restrict__ VpT,
    float* __restrict__ Of, bf16_t* __restrict__ Obf,
    int Bn, int SQn, int SKn, int H, int DV, float scale) {
  __shared__ bf16_t pbuf[4][16 * 32];

  const int lane = threadIdx.x & 31;
  const int wave = threadIdx.x >> 5;
  const int qTiles = SQn >> 4;
  const int task = blockIdx.x * 4 + wave;
  const int qt = task % qTiles;
  const int bh = task / qTiles;
  const int h = bh % H;
  const int b = bh / H;
  const int m0 = qt << 4;
  const int row = lane & 15;
  const int hi = lane >> 4;
  const int dh = DV / H;  // 64

  // Q head-slice as two 16x32 A fragments (d = 0..31, 32..63)
  const size_t qrow = (size_t)(b * SQn + m0 + row) * DV + h * dh;
  v16bf aq[2];
#pragma unroll
  for (int t = 0; t < 2; ++t) {
    const bf16_t* p = Qbf + qrow + t * 32 + hi * 16;
    aq[t] = join16(*(const v8bf*)p, *(const v8bf*)(p + 8));
  }

  v8f ctx[4] = {v8f{}, v8f{}, v8f{}, v8f{}};
  float mrun[8], lrun[8];
#pragma unroll
  for (int r = 0; r < 8; ++r) {
    mrun[r] = -__builtin_inff();
    lrun[r] = 0.0f;
  }

  bf16_t* pt = pbuf[wave];
  // K rows for this lane's fragment column; advance by 32 keys per iter.
  const bf16_t* kb = Kbf + (size_t)(b * SKn + row) * DV + h * dh + hi * 16;
  const size_t kRow16 = (size_t)16 * DV;
  // transposed V: [b][h][d][sk]; fragment j column d = j*16+row, k contiguous.
  const bf16_t* vt = VpT + ((size_t)((b * H + h) * dh) + row) * SKn + hi * 16;
  const size_t vCol16 = (size_t)16 * SKn;

  for (int kt = 0; kt < SKn; kt += 32) {
    // ---- scores: S(16x32) = Qh(16x64) @ Kh^T ------------------------------
    v8f s0 = {}, s1 = {};
    s0 = WMMA_BF16(aq[0], *(const v16bf*)(kb), s0);
    s0 = WMMA_BF16(aq[1], *(const v16bf*)(kb + 32), s0);
    s1 = WMMA_BF16(aq[0], *(const v16bf*)(kb + kRow16), s1);
    s1 = WMMA_BF16(aq[1], *(const v16bf*)(kb + kRow16 + 32), s1);

    // ---- online softmax ---------------------------------------------------
    float p0[8], p1[8], cf[8];
#pragma unroll
    for (int r = 0; r < 8; ++r) {
      const float a0 = s0[r] * scale;
      const float a1 = s1[r] * scale;
      const float mx = xmax16(fmaxf(a0, a1));
      const float mnew = fmaxf(mrun[r], mx);
      cf[r] = __expf(mrun[r] - mnew);
      p0[r] = __expf(a0 - mnew);
      p1[r] = __expf(a1 - mnew);
      lrun[r] = lrun[r] * cf[r] + xsum16(p0[r] + p1[r]);
      mrun[r] = mnew;
    }
#pragma unroll
    for (int j = 0; j < 4; ++j)
#pragma unroll
      for (int r = 0; r < 8; ++r) ctx[j][r] *= cf[r];

    // ---- restage P (C-layout) through LDS into A-fragment layout ----------
#pragma unroll
    for (int r = 0; r < 8; ++r) {
      pt[(hi * 8 + r) * 32 + row] = (bf16_t)p0[r];
      pt[(hi * 8 + r) * 32 + 16 + row] = (bf16_t)p1[r];
    }
    __syncthreads();
    const bf16_t* pr = pt + row * 32 + hi * 16;
    v16bf ap = join16(*(const v8bf*)pr, *(const v8bf*)(pr + 8));

    // ---- ctx += P(16x32) @ V(32x64), V transposed -> contiguous loads -----
#pragma unroll
    for (int j = 0; j < 4; ++j) {
      v16bf bv = *(const v16bf*)(vt + (size_t)j * vCol16);
      ctx[j] = WMMA_BF16(ap, bv, ctx[j]);
    }
    __syncthreads();

    kb += (size_t)32 * DV;
    vt += 32;
  }

  // ---- epilogue: O = Qh + ctx/l ------------------------------------------
  float invl[8];
#pragma unroll
  for (int r = 0; r < 8; ++r) invl[r] = 1.0f / lrun[r];
#pragma unroll
  for (int j = 0; j < 4; ++j) {
#pragma unroll
    for (int r = 0; r < 8; ++r) {
      const size_t idx =
          (size_t)(b * SQn + m0 + hi * 8 + r) * DV + h * dh + j * 16 + row;
      const float v = Qf[idx] + ctx[j][r] * invl[r];
      Of[idx] = v;
      Obf[idx] = (bf16_t)v;
    }
  }
}

// ---------------------------------------------------------------------------
// Host-side orchestration
// ---------------------------------------------------------------------------
extern "C" void kernel_launch(void* const* d_in, const int* in_sizes, int n_in,
                              void* d_out, int out_size, void* d_ws,
                              size_t ws_size, hipStream_t stream) {
  constexpr int Bb = 8, SQ = 1024, SK = 1024, D = 512, H = 8;
  constexpr int M = Bb * SQ;  // 8192 (== Bb*SK)

  const float* Q = (const float*)d_in[0];
  const float* K = (const float*)d_in[1];
  const float* Wq = (const float*)d_in[2];
  const float* bq = (const float*)d_in[3];
  const float* Wk = (const float*)d_in[4];
  const float* bk = (const float*)d_in[5];
  const float* Wv = (const float*)d_in[6];
  const float* bv = (const float*)d_in[7];
  const float* Wo = (const float*)d_in[8];
  const float* bo = (const float*)d_in[9];

  // workspace carve-up
  char* base = (char*)d_ws;
  size_t off = 0;
  auto carve = [&](size_t bytes) {
    void* p = base + off;
    off = (off + bytes + 255) & ~(size_t)255;
    return p;
  };
  const size_t actBytesBf = (size_t)M * D * sizeof(bf16_t);
  const size_t actBytesF = (size_t)M * D * sizeof(float);
  const size_t wBytesBf = (size_t)D * D * sizeof(bf16_t);

  bf16_t* Qin_bf = (bf16_t*)carve(actBytesBf);
  bf16_t* Kin_bf = (bf16_t*)carve(actBytesBf);
  bf16_t* Wq_bf = (bf16_t*)carve(wBytesBf);
  bf16_t* Wk_bf = (bf16_t*)carve(wBytesBf);
  bf16_t* Wv_bf = (bf16_t*)carve(wBytesBf);
  bf16_t* Wo_bf = (bf16_t*)carve(wBytesBf);
  float* Qp_f = (float*)carve(actBytesF);
  bf16_t* Qp_bf = (bf16_t*)carve(actBytesBf);
  bf16_t* Kp_bf = (bf16_t*)carve(actBytesBf);
  bf16_t* VpT_bf = (bf16_t*)carve(actBytesBf);  // [B][H][dh][SK]
  float* Oi_f = (float*)carve(actBytesF);
  bf16_t* Oi_bf = (bf16_t*)carve(actBytesBf);

  // 1) fp32 -> bf16 conversions (4-wide)
  const int nAct4 = M * D / 4, nW4 = D * D / 4;
  cvt_f32_to_bf16<<<(nAct4 + 255) / 256, 256, 0, stream>>>(Q, Qin_bf, nAct4);
  cvt_f32_to_bf16<<<(nAct4 + 255) / 256, 256, 0, stream>>>(K, Kin_bf, nAct4);
  cvt_f32_to_bf16<<<(nW4 + 255) / 256, 256, 0, stream>>>(Wq, Wq_bf, nW4);
  cvt_f32_to_bf16<<<(nW4 + 255) / 256, 256, 0, stream>>>(Wk, Wk_bf, nW4);
  cvt_f32_to_bf16<<<(nW4 + 255) / 256, 256, 0, stream>>>(Wv, Wv_bf, nW4);
  cvt_f32_to_bf16<<<(nW4 + 255) / 256, 256, 0, stream>>>(Wo, Wo_bf, nW4);

  // 2) projections
  const int gemmBlocks = (M / 16) * (D / 64) / 4;  // 1024
  gemm_bf16_wmma<0><<<gemmBlocks, 128, 0, stream>>>(
      Qin_bf, Wq_bf, bq, nullptr, Qp_f, Qp_bf, M, D, D, SK);
  gemm_bf16_wmma<3><<<gemmBlocks, 128, 0, stream>>>(
      Kin_bf, Wk_bf, bk, nullptr, nullptr, Kp_bf, M, D, D, SK);
  gemm_bf16_wmma<2><<<gemmBlocks, 128, 0, stream>>>(
      Kin_bf, Wv_bf, bv, nullptr, nullptr, VpT_bf, M, D, D, SK);

  // 3) flash attention + Q residual -> Oi
  const float scale = 0.044194173824159216f;  // 1/sqrt(512)
  const int attnBlocks = Bb * H * (SQ / 16) / 4;  // 1024
  attention_flash<<<attnBlocks, 128, 0, stream>>>(
      Qp_bf, Qp_f, Kp_bf, VpT_bf, Oi_f, Oi_bf, Bb, SQ, SK, H, D, scale);

  // 4) FFN: out = Oi + relu(Oi @ Wo^T + bo)
  gemm_bf16_wmma<1><<<gemmBlocks, 128, 0, stream>>>(
      Oi_bf, Wo_bf, bo, Oi_f, (float*)d_out, nullptr, M, D, D, SK);
}
